// SpGQAFlashDecodeAttentionMPI_21225728376897
// MI455X (gfx1250) — matrix operand order repaired and checked by reading, not compiled, not run on hardware
//
#include <hip/hip_runtime.h>
#include <math.h>

typedef _Float16 f16;
typedef __attribute__((ext_vector_type(16))) _Float16 v16h;
typedef __attribute__((ext_vector_type(8)))  float    v8f;

#define R_      4
#define B_      4
#define HQ_     32
#define HK_     8
#define G_      4
#define D_      128
#define DV_     128
#define N_      4096
#define PAGES_  (B_ * N_)
#define SPLITS_ 4
#define WAVES_  4
#define SCALE_  0.08838834764831845f   /* 1/sqrt(128) */

/* LDS layout (bytes). Padded strides (in halves) chosen for bank spread. */
#define KSTRIDE 136                    /* 128 + 8 halves  -> token rows hit distinct banks */
#define VSTRIDE 136
#define PSTRIDE 34                     /* 32 + 2 halves */
#define QS_BYTES   (16 * D_ * 2)                       /* 4096  */
#define KT_BYTES   (32 * KSTRIDE * 2)                  /* 8704  */
#define VT_BYTES   (32 * VSTRIDE * 2)                  /* 8704  */
#define PT_BYTES   (16 * PSTRIDE * 2)                  /* 1088  */
#define PER_WAVE_BYTES (KT_BYTES + VT_BYTES + PT_BYTES)/* 18496 */
#define COMB_O_OFF (QS_BYTES + WAVES_ * PER_WAVE_BYTES)/* 78080 */
#define COMB_ML_OFF (COMB_O_OFF + WAVES_ * G_ * DV_ * 4)/* 86272 */
#define SMEM_BYTES (COMB_ML_OFF + WAVES_ * 2 * G_ * 4)  /* 86400 */

__global__ __launch_bounds__(WAVES_ * 32) void
SpGQAFlashDecode_partial_kernel(const float* __restrict__ q,
                                const float* __restrict__ kc,
                                const float* __restrict__ vc,
                                const int*   __restrict__ lens,
                                const int*   __restrict__ bt,
                                float* __restrict__ pout,
                                float* __restrict__ plse)
{
    extern __shared__ char smem[];
    f16* qs = (f16*)smem;

    const int tid  = threadIdx.x;
    const int lane = tid & 31;
    const int w    = tid >> 5;

    int idx = blockIdx.x;                       /* (((r*B + b)*HK + hk)*SPLITS + split) */
    const int split = idx & (SPLITS_ - 1); idx >>= 2;
    const int hk    = idx & (HK_ - 1);     idx >>= 3;
    const int b     = idx & (B_ - 1);      idx >>= 2;
    const int r     = idx;

    const int len     = lens[r * B_ + b];
    const int chunk   = (len + SPLITS_ - 1) / SPLITS_;
    const int t_begin = split * chunk;
    const int t_end   = min(len, t_begin + chunk);

    /* ---- stage q (scaled) into LDS as f16, rows G..15 zeroed ---- */
    {
        const int row = tid >> 3;
        const int db  = (tid & 7) * 16;
        const float* qrow = q + ((size_t)b * HQ_ + hk * G_ + (row < G_ ? row : 0)) * D_;
        #pragma unroll
        for (int i = 0; i < 16; ++i) {
            float v = (row < G_) ? qrow[db + i] * SCALE_ : 0.0f;
            qs[row * D_ + db + i] = (f16)v;
        }
    }

    f16* kt = (f16*)(smem + QS_BYTES + w * PER_WAVE_BYTES);
    f16* vt = kt + 32 * KSTRIDE;
    f16* pt = vt + 32 * VSTRIDE;
    float* combO = (float*)(smem + COMB_O_OFF);   /* [W][G][DV] */
    float* combM = (float*)(smem + COMB_ML_OFF);  /* [W][{m,l}][G] packed as w*8 + {0..3,4..7} */

    /* zero P staging (rows >= G stay zero forever -> WMMA rows 4..15 are null) */
    for (int i = lane; i < 16 * PSTRIDE; i += 32) pt[i] = (f16)0.0f;
    __syncthreads();

    const int rowm = lane & 15;
    const int koff = (lane & 16) ? 8 : 0;

    /* ---- prebuild A(q) tiles: 4 chunks of K=32 along D ---- */
    v16h aq[4];
    #pragma unroll
    for (int c = 0; c < 4; ++c) {
        #pragma unroll
        for (int j = 0; j < 16; ++j) {
            int kd = c * 32 + (j & 7) + koff + ((j >> 3) << 4);
            aq[c][j] = qs[rowm * D_ + kd];
        }
    }

    const v8f zero8 = {0.f, 0.f, 0.f, 0.f, 0.f, 0.f, 0.f, 0.f};
    v8f o[8];
    #pragma unroll
    for (int k = 0; k < 8; ++k) o[k] = zero8;
    float m[G_], l[G_];
    #pragma unroll
    for (int g = 0; g < G_; ++g) { m[g] = -INFINITY; l[g] = 0.0f; }

    const int*   btb   = bt + ((size_t)r * B_ + b) * N_;
    const size_t kbase = (size_t)r * PAGES_ * HK_ * D_ + (size_t)hk * D_;

    for (int t0 = t_begin + w * 32; t0 < t_end; t0 += WAVES_ * 32) {
        /* ---- stage 32 tokens of K,V (fp32 -> f16) with b128 loads ---- */
        const int t  = t0 + lane;
        const int tc = (t < t_end) ? t : (t_end - 1);
        const int page = btb[tc];
        const float* kp = kc + kbase + (size_t)page * (HK_ * D_);
        const float* vp = vc + kbase + (size_t)page * (HK_ * DV_);
        #pragma unroll
        for (int c4 = 0; c4 < 32; ++c4) {
            float4 k4 = ((const float4*)kp)[c4];
            float4 v4 = ((const float4*)vp)[c4];
            f16* kd = kt + lane * KSTRIDE + c4 * 4;
            kd[0] = (f16)k4.x; kd[1] = (f16)k4.y; kd[2] = (f16)k4.z; kd[3] = (f16)k4.w;
            f16* vd = vt + lane * VSTRIDE + c4 * 4;
            vd[0] = (f16)v4.x; vd[1] = (f16)v4.y; vd[2] = (f16)v4.z; vd[3] = (f16)v4.w;
        }
        /* prefetch next tile (global_prefetch_b8) */
        if (t0 + WAVES_ * 32 < t_end) {
            int tn = t0 + WAVES_ * 32 + lane;
            tn = (tn < t_end) ? tn : (t_end - 1);
            const int pn = btb[tn];
            __builtin_prefetch(kc + kbase + (size_t)pn * (HK_ * D_), 0, 1);
            __builtin_prefetch(vc + kbase + (size_t)pn * (HK_ * DV_), 0, 1);
        }
        __builtin_amdgcn_wave_barrier();   /* order stage stores before WMMA operand reads */

        /* ---- QK: S = q @ K^T, two 16-token column tiles, K-dim = 128 ---- */
        v8f sa = zero8, sb = zero8;
        #pragma unroll
        for (int c = 0; c < 4; ++c) {
            v16h bk0, bk1;
            #pragma unroll
            for (int j = 0; j < 16; ++j) {
                int kd = c * 32 + (j & 7) + koff + ((j >> 3) << 4);
                bk0[j] = kt[rowm * KSTRIDE + kd];
                bk1[j] = kt[(16 + rowm) * KSTRIDE + kd];
            }
            sa = __builtin_amdgcn_wmma_f32_16x16x32_f16(false, aq[c], false, bk0,
                                                        (short)0, sa, false, false);
            sb = __builtin_amdgcn_wmma_f32_16x16x32_f16(false, aq[c], false, bk1,
                                                        (short)0, sb, false, false);
        }

        /* ---- online softmax over the 32-token group (per q row g) ---- */
        const int tcol_a = t0 + rowm;
        const int tcol_b = t0 + 16 + rowm;
        #pragma unroll
        for (int g = 0; g < G_; ++g) {
            float s0 = (tcol_a < t_end) ? sa[g] : -INFINITY;
            float s1 = (tcol_b < t_end) ? sb[g] : -INFINITY;
            float tmax = fmaxf(s0, s1);
            #pragma unroll
            for (int off = 1; off < 16; off <<= 1)
                tmax = fmaxf(tmax, __shfl_xor(tmax, off, 32));
            const float nm = fmaxf(m[g], tmax);
            const bool ok  = (nm > -INFINITY);
            const float alpha = ok ? __expf(m[g] - nm) : 1.0f;
            const float p0 = ok ? __expf(s0 - nm) : 0.0f;
            const float p1 = ok ? __expf(s1 - nm) : 0.0f;
            float ps = p0 + p1;
            #pragma unroll
            for (int off = 1; off < 16; off <<= 1)
                ps += __shfl_xor(ps, off, 32);
            l[g] = l[g] * alpha + ps;
            m[g] = nm;
            if (lane < 16) {
                pt[g * PSTRIDE + rowm]      = (f16)p0;
                pt[g * PSTRIDE + 16 + rowm] = (f16)p1;
            }
            #pragma unroll
            for (int k = 0; k < 8; ++k) o[k][g] = o[k][g] * alpha;
        }
        __builtin_amdgcn_wave_barrier();

        /* ---- PV: O += P @ V, K-dim = 32 tokens, 8 dv chunks of 16 ---- */
        v16h ap;
        #pragma unroll
        for (int j = 0; j < 16; ++j) {
            int tk = (j & 7) + koff + ((j >> 3) << 4);
            ap[j] = pt[rowm * PSTRIDE + tk];
        }
        #pragma unroll
        for (int dc = 0; dc < 8; ++dc) {
            v16h bv;
            #pragma unroll
            for (int j = 0; j < 16; ++j) {
                int tk = (j & 7) + koff + ((j >> 3) << 4);
                bv[j] = vt[tk * VSTRIDE + dc * 16 + rowm];
            }
            o[dc] = __builtin_amdgcn_wmma_f32_16x16x32_f16(false, ap, false, bv,
                                                           (short)0, o[dc], false, false);
        }
    }

    /* ---- per-wave partials to LDS ---- */
    if (lane < 16) {
        #pragma unroll
        for (int g = 0; g < G_; ++g)
            #pragma unroll
            for (int dc = 0; dc < 8; ++dc)
                combO[(w * G_ + g) * DV_ + dc * 16 + rowm] = o[dc][g];
    }
    if (lane == 0) {
        #pragma unroll
        for (int g = 0; g < G_; ++g) {
            combM[w * 8 + g]      = m[g];
            combM[w * 8 + 4 + g]  = l[g];
        }
    }
    __syncthreads();

    /* ---- cross-wave merge -> normalized partial + LSE for this (r,split,b,hk) ---- */
    {
        const int g   = tid >> 5;
        const int sub = tid & 31;
        float M = -INFINITY;
        #pragma unroll
        for (int ww = 0; ww < WAVES_; ++ww) M = fmaxf(M, combM[ww * 8 + g]);
        float ls = 0.0f, acc[4] = {0.f, 0.f, 0.f, 0.f};
        #pragma unroll
        for (int ww = 0; ww < WAVES_; ++ww) {
            const float mw = combM[ww * 8 + g];
            const float a  = (M > -INFINITY && mw > -INFINITY) ? __expf(mw - M) : 0.0f;
            ls += a * combM[ww * 8 + 4 + g];
            #pragma unroll
            for (int i = 0; i < 4; ++i)
                acc[i] += a * combO[(ww * G_ + g) * DV_ + sub * 4 + i];
        }
        const int h    = hk * G_ + g;
        const int pidx = ((r * SPLITS_ + split) * B_ + b) * HQ_ + h;
        const float inv = (ls > 0.0f) ? 1.0f / ls : 0.0f;
        #pragma unroll
        for (int i = 0; i < 4; ++i)
            pout[(size_t)pidx * DV_ + sub * 4 + i] = acc[i] * inv;
        if (sub == 0)
            plse[pidx] = (ls > 0.0f) ? (M + __logf(ls)) : -INFINITY;
    }
}

/* ---- inter-rank / inter-split LSE-weighted combine ---- */
__global__ __launch_bounds__(DV_) void
SpGQAFlashDecode_combine_kernel(const float* __restrict__ pout,
                                const float* __restrict__ plse,
                                float* __restrict__ out)
{
    const int bh = blockIdx.x;          /* b*HQ + h */
    const int dv = threadIdx.x;
    float M = -INFINITY;
    #pragma unroll
    for (int p = 0; p < R_ * SPLITS_; ++p)
        M = fmaxf(M, plse[(size_t)p * (B_ * HQ_) + bh]);
    float den = 0.0f, acc = 0.0f;
    #pragma unroll
    for (int p = 0; p < R_ * SPLITS_; ++p) {
        const float lp = plse[(size_t)p * (B_ * HQ_) + bh];
        const float wt = (lp > -INFINITY) ? __expf(lp - M) : 0.0f;
        den += wt;
        acc += wt * pout[((size_t)p * (B_ * HQ_) + bh) * DV_ + dv];
    }
    out[(size_t)bh * DV_ + dv] = acc / den;
}

extern "C" void kernel_launch(void* const* d_in, const int* in_sizes, int n_in,
                              void* d_out, int out_size, void* d_ws, size_t ws_size,
                              hipStream_t stream) {
    (void)in_sizes; (void)n_in; (void)out_size; (void)ws_size;
    const float* q    = (const float*)d_in[0];
    const float* kc   = (const float*)d_in[1];
    const float* vc   = (const float*)d_in[2];
    const int*   lens = (const int*)d_in[3];
    const int*   bt   = (const int*)d_in[4];

    float* pout = (float*)d_ws;                                   /* [R*SPLITS][B][HQ][DV] */
    float* plse = pout + (size_t)R_ * SPLITS_ * B_ * HQ_ * DV_;   /* [R*SPLITS][B][HQ]     */

    SpGQAFlashDecode_partial_kernel<<<R_ * B_ * HK_ * SPLITS_, WAVES_ * 32,
                                      SMEM_BYTES, stream>>>(q, kc, vc, lens, bt, pout, plse);
    SpGQAFlashDecode_combine_kernel<<<B_ * HQ_, DV_, 0, stream>>>(pout, plse, (float*)d_out);
}
